// PVCNN2Base_42545946034414
// MI455X (gfx1250) — matrix operand order repaired and checked by reading, not compiled, not run on hardware
//
#include <hip/hip_runtime.h>

// ---------------------------------------------------------------------------
// PVCNN forward for MI455X (gfx1250, wave32, WMMA 16x16x32 f16 -> f32)
// Round 2: pre-converted f16 weights (no per-wave cvt chains), K/V prefetch.
// ---------------------------------------------------------------------------

typedef __attribute__((ext_vector_type(16))) _Float16 v16h;
typedef __attribute__((ext_vector_type(8)))  _Float16 v8h;
typedef __attribute__((ext_vector_type(8)))  float    v8f;

#define BB    4
#define NN    32768
#define CIN   6
#define CC    64
#define RR    16
#define SSZ   4096
#define MLPH  128
#define NCLS  3
#define SEH   8

// ---------------- small device helpers ----------------

__device__ __forceinline__ v8f vzero8() {
  v8f z;
#pragma unroll
  for (int i = 0; i < 8; ++i) z[i] = 0.f;
  return z;
}

// D = A(16x32 f16) * B(32x16 f16) + C  (f32 accum)
__device__ __forceinline__ v8f wmma16(v16h a, v16h b, v8f c) {
  return __builtin_amdgcn_wmma_f32_16x16x32_f16(false, a, false, b, (short)0, c, false, false);
}

// A-operand (16x32, f16) from a row-major f16 row pointer (row = this lane's M row,
// already offset to the 32-column chunk). Layout: element e -> col
// (lane/16)*16 + e  => one contiguous 16-half run.
__device__ __forceinline__ v16h load_a_f16(const _Float16* rowChunk, int half) {
  const v8h lo = *(const v8h*)(rowChunk + half * 16);
  const v8h hi = *(const v8h*)(rowChunk + half * 16 + 8);
  v16h r;
#pragma unroll
  for (int i = 0; i < 8; ++i) { r[i] = lo[i]; r[i + 8] = hi[i]; }
  return r;
}

// Row-wise reductions: a score row lives in one 16-lane half (C/D layout),
// xor masks 1/2/4/8 never cross the 16-lane boundary in wave32.
__device__ __forceinline__ float red16_max(float v) {
  v = fmaxf(v, __shfl_xor(v, 1));
  v = fmaxf(v, __shfl_xor(v, 2));
  v = fmaxf(v, __shfl_xor(v, 4));
  v = fmaxf(v, __shfl_xor(v, 8));
  return v;
}
__device__ __forceinline__ float red16_sum(float v) {
  v += __shfl_xor(v, 1);
  v += __shfl_xor(v, 2);
  v += __shfl_xor(v, 4);
  v += __shfl_xor(v, 8);
  return v;
}

__device__ __forceinline__ void lds_fence() {
  asm volatile("s_wait_dscnt 0" ::: "memory");
}

// ---------------- kernels ----------------

__global__ void pvk_zero(float* p, int n) {
  int i = blockIdx.x * 256 + threadIdx.x;
  if (i < n) p[i] = 0.f;
}

__global__ void pvk_f32tof16(const float* __restrict__ src, _Float16* __restrict__ dst, int n) {
  int i = blockIdx.x * 256 + threadIdx.x;
  if (i < n) dst[i] = (_Float16)src[i];
}

// scatter-mean voxelize (accumulate)
__global__ void pvk_scatter(const float* __restrict__ coords,
                            const float* __restrict__ feats,
                            float* __restrict__ gsum, float* __restrict__ gcnt) {
  int idx = blockIdx.x * 256 + threadIdx.x;       // B*N
  int b = idx / NN, n = idx % NN;
  const float* cb = coords + (size_t)b * 3 * NN;
  int vx = (int)floorf(cb[n] * (float)RR);
  int vy = (int)floorf(cb[NN + n] * (float)RR);
  int vz = (int)floorf(cb[2 * NN + n] * (float)RR);
  vx = vx < 0 ? 0 : (vx > RR - 1 ? RR - 1 : vx);
  vy = vy < 0 ? 0 : (vy > RR - 1 ? RR - 1 : vy);
  vz = vz < 0 ? 0 : (vz > RR - 1 ? RR - 1 : vz);
  int cell = (vx * RR + vy) * RR + vz;
  atomicAdd(&gcnt[b * SSZ + cell], 1.f);
#pragma unroll
  for (int i = 0; i < CIN; ++i)
    atomicAdd(&gsum[((size_t)b * CIN + i) * SSZ + cell],
              feats[((size_t)b * CIN + i) * NN + n]);
}

__global__ void pvk_mean(float* __restrict__ gsum, const float* __restrict__ gcnt) {
  int idx = blockIdx.x * 256 + threadIdx.x;       // B*CIN*S
  int b = idx / (CIN * SSZ);
  int cell = idx % SSZ;
  gsum[idx] = gsum[idx] / fmaxf(gcnt[b * SSZ + cell], 1.f);
}

// 3x3x3 conv (SAME, zero pad) + ReLU; write f32 [b][c][s] and f16 [b][s][c]
__global__ void pvk_conv(const float* __restrict__ gmean,
                         const float* __restrict__ conv_w,
                         const float* __restrict__ conv_b,
                         float* __restrict__ x_cs, _Float16* __restrict__ xT) {
  int idx = blockIdx.x * 256 + threadIdx.x;       // B*C*S
  int b = idx / (CC * SSZ);
  int rem = idx % (CC * SSZ);
  int o = rem / SSZ, cell = rem % SSZ;
  int xg = cell >> 8, yg = (cell >> 4) & 15, zg = cell & 15;
  float acc = conv_b[o];
  for (int i = 0; i < CIN; ++i) {
    const float* g = gmean + ((size_t)b * CIN + i) * SSZ;
    const float* w = conv_w + ((size_t)o * CIN + i) * 27;
#pragma unroll
    for (int dx = 0; dx < 3; ++dx) {
      int xx = xg + dx - 1; if ((unsigned)xx >= (unsigned)RR) continue;
#pragma unroll
      for (int dy = 0; dy < 3; ++dy) {
        int yy = yg + dy - 1; if ((unsigned)yy >= (unsigned)RR) continue;
#pragma unroll
        for (int dz = 0; dz < 3; ++dz) {
          int zz = zg + dz - 1; if ((unsigned)zz >= (unsigned)RR) continue;
          acc += w[dx * 9 + dy * 3 + dz] * g[(xx << 8) + (yy << 4) + zz];
        }
      }
    }
  }
  acc = fmaxf(acc, 0.f);
  x_cs[((size_t)b * CC + o) * SSZ + cell] = acc;
  xT[((size_t)b * SSZ + cell) * CC + o] = (_Float16)acc;
}

// q,k,v = Wq/Wk/Wv (64x64) * x(64xS).  One wave per 16-column s-tile.
// qT,kT: [b][s][c] f16 ; v: [b][c][s] f16
__global__ void pvk_qkv(const _Float16* __restrict__ wqh, const _Float16* __restrict__ wkh,
                        const _Float16* __restrict__ wvh, const _Float16* __restrict__ xT,
                        _Float16* __restrict__ qT, _Float16* __restrict__ kT,
                        _Float16* __restrict__ v_cs) {
  const int lane = threadIdx.x & 31, wave = threadIdx.x >> 5;
  const int tile = blockIdx.x * 8 + wave;         // B*S/16 tiles
  const int b = tile >> 8, s0 = (tile & 255) << 4;
  const int half = lane >> 4, ln = lane & 15;

  // B operand: x chunk (c x 16s); element e -> c = cc*32 + half*16 + e
  const _Float16* xrow = xT + ((size_t)b * SSZ + s0 + ln) * CC;
  v16h xb0 = *(const v16h*)(xrow + half * 16);
  v16h xb1 = *(const v16h*)(xrow + 32 + half * 16);

  for (int mat = 0; mat < 3; ++mat) {
    const _Float16* W = (mat == 0) ? wqh : ((mat == 1) ? wkh : wvh);
#pragma unroll
    for (int ot = 0; ot < 4; ++ot) {
      const _Float16* wr = W + (size_t)(ot * 16 + ln) * CC;
      v16h a0 = load_a_f16(wr, half);
      v16h a1 = load_a_f16(wr + 32, half);
      v8f acc = vzero8();
      acc = wmma16(a0, xb0, acc);
      acc = wmma16(a1, xb1, acc);
      if (mat == 2) {
#pragma unroll
        for (int r = 0; r < 8; ++r) {
          int o = ot * 16 + r + half * 8;
          v_cs[((size_t)b * CC + o) * SSZ + s0 + ln] = (_Float16)acc[r];
        }
      } else {
        _Float16* dst = ((mat == 0) ? qT : kT) +
                        ((size_t)b * SSZ + s0 + ln) * CC + ot * 16 + half * 8;
#pragma unroll
        for (int r = 0; r < 8; ++r) dst[r] = (_Float16)acc[r];
      }
    }
  }
}

// Flash attention over S=4096 cells, d=64, scale 1/8. One wave per 16 rows.
__global__ void pvk_flash(const _Float16* __restrict__ qT,
                          const _Float16* __restrict__ kT,
                          const _Float16* __restrict__ v_cs,
                          _Float16* __restrict__ pv) {
  __shared__ __align__(64) _Float16 s_P[8][16 * 32];

  const int lane = threadIdx.x & 31, wave = threadIdx.x >> 5;
  const int tile = blockIdx.x * 8 + wave;
  const int b = tile >> 8, s0 = (tile & 255) << 4;
  const int half = lane >> 4, ln = lane & 15;

  const _Float16* qTb = qT + (size_t)b * SSZ * CC;
  const _Float16* kTb = kT + (size_t)b * SSZ * CC;
  const _Float16* vb  = v_cs + (size_t)b * CC * SSZ;
  _Float16* pvb = pv + (size_t)b * SSZ * CC;
  _Float16* Pbuf = &s_P[wave][0];

  // Q tile as two A operands (c 0..31 / 32..63)
  const _Float16* qrow = qTb + (size_t)(s0 + ln) * CC;
  v16h qa0 = load_a_f16(qrow, half);
  v16h qa1 = load_a_f16(qrow + 32, half);

  v8f oacc[4];
#pragma unroll
  for (int i = 0; i < 4; ++i) oacc[i] = vzero8();
  float mrow[8], lrow[8];
#pragma unroll
  for (int r = 0; r < 8; ++r) { mrow[r] = -1e30f; lrow[r] = 0.f; }

  for (int t0 = 0; t0 < SSZ; t0 += 32) {
    // prefetch next iteration's K/V tiles into cache (global_prefetch_b8)
    if (t0 + 32 < SSZ) {
      __builtin_prefetch(kTb + (size_t)(t0 + 32 + lane) * CC, 0, 1);
      __builtin_prefetch(vb + (size_t)lane * SSZ + t0 + 32, 0, 1);
      __builtin_prefetch(vb + (size_t)(32 + lane) * SSZ + t0 + 32, 0, 1);
    }

    // two 16x16 score tiles
    v8f sc[2];
#pragma unroll
    for (int tt = 0; tt < 2; ++tt) {
      const _Float16* krow = kTb + (size_t)(t0 + tt * 16 + ln) * CC;
      v16h kb0 = *(const v16h*)(krow + half * 16);
      v16h kb1 = *(const v16h*)(krow + 32 + half * 16);
      v8f a = vzero8();
      a = wmma16(qa0, kb0, a);
      a = wmma16(qa1, kb1, a);
      sc[tt] = a;
    }

    // online softmax over these 32 columns
    float alpha[8];
#pragma unroll
    for (int r = 0; r < 8; ++r) {
      float x0 = sc[0][r] * 0.125f;
      float x1 = sc[1][r] * 0.125f;
      float rm = red16_max(fmaxf(x0, x1));
      float mn = fmaxf(mrow[r], rm);
      float a  = __expf(mrow[r] - mn);
      float p0 = __expf(x0 - mn);
      float p1 = __expf(x1 - mn);
      lrow[r] = lrow[r] * a + red16_sum(p0 + p1);
      mrow[r] = mn;
      alpha[r] = a;
      int m = r + half * 8;                        // score row of this element
      Pbuf[m * 32 + ln]      = (_Float16)p0;
      Pbuf[m * 32 + 16 + ln] = (_Float16)p1;
    }
#pragma unroll
    for (int ct = 0; ct < 4; ++ct)
#pragma unroll
      for (int r = 0; r < 8; ++r) oacc[ct][r] *= alpha[r];

    lds_fence();                                   // same-wave LDS RAW

    // P (16s x 32t) re-read as A operand
    v16h pa = load_a_f16(Pbuf + ln * 32, half);

    // O += P * V^T : B operand = v[c][t] chunk (32t x 16c)
#pragma unroll
    for (int ct = 0; ct < 4; ++ct) {
      const _Float16* vrow = vb + (size_t)(ct * 16 + ln) * SSZ + t0 + half * 16;
      v16h vbop = *(const v16h*)vrow;
      oacc[ct] = wmma16(pa, vbop, oacc[ct]);
    }
  }

  // normalize and store pv [s][c] f16
#pragma unroll
  for (int ct = 0; ct < 4; ++ct)
#pragma unroll
    for (int r = 0; r < 8; ++r) {
      float o = oacc[ct][r] / lrow[r];
      int s = s0 + r + half * 8;
      pvb[(size_t)s * CC + ct * 16 + ln] = (_Float16)o;
    }
}

// x2 = x + Wo*pv ; accumulate channel sums for SE.
__global__ void pvk_attnout(const _Float16* __restrict__ woh,
                            const _Float16* __restrict__ pv,
                            const float* __restrict__ x_cs,
                            float* __restrict__ x2, float* __restrict__ se_sum) {
  const int lane = threadIdx.x & 31, wave = threadIdx.x >> 5;
  const int tile = blockIdx.x * 8 + wave;
  const int b = tile >> 8, s0 = (tile & 255) << 4;
  const int half = lane >> 4, ln = lane & 15;

  const _Float16* prow = pv + ((size_t)b * SSZ + s0 + ln) * CC;
  v16h b0 = *(const v16h*)(prow + half * 16);
  v16h b1 = *(const v16h*)(prow + 32 + half * 16);

#pragma unroll
  for (int ot = 0; ot < 4; ++ot) {
    const _Float16* wr = woh + (size_t)(ot * 16 + ln) * CC;
    v16h a0 = load_a_f16(wr, half);
    v16h a1 = load_a_f16(wr + 32, half);
    v8f acc = vzero8();
    acc = wmma16(a0, b0, acc);
    acc = wmma16(a1, b1, acc);
#pragma unroll
    for (int r = 0; r < 8; ++r) {
      int o = ot * 16 + r + half * 8;
      size_t idx = ((size_t)b * CC + o) * SSZ + s0 + ln;
      float v = x_cs[idx] + acc[r];
      x2[idx] = v;
      float rs = red16_sum(v);
      if (ln == 0) atomicAdd(&se_sum[b * CC + o], rs);
    }
  }
}

// squeeze-excite gate: sigmoid(W2 relu(W1 mean + b1) + b2)
__global__ void pvk_gate(const float* __restrict__ se_sum,
                         const float* __restrict__ w1, const float* __restrict__ b1,
                         const float* __restrict__ w2, const float* __restrict__ b2,
                         float* __restrict__ gate) {
  int t = threadIdx.x;                            // 256 = B*C
  int b = t >> 6, c = t & 63;
  float h[SEH];
#pragma unroll
  for (int hh = 0; hh < SEH; ++hh) {
    float a = b1[hh];
    for (int j = 0; j < CC; ++j)
      a += w1[hh * CC + j] * (se_sum[b * CC + j] * (1.f / (float)SSZ));
    h[hh] = fmaxf(a, 0.f);
  }
  float g = b2[c];
#pragma unroll
  for (int hh = 0; hh < SEH; ++hh) g += w2[c * SEH + hh] * h[hh];
  gate[b * CC + c] = 1.f / (1.f + __expf(-g));
}

// devoxelize (trilinear, gated) + point MLP + fuse + classifier (WMMA for W1)
__global__ void pvk_point(const float* __restrict__ coords,
                          const float* __restrict__ feats,
                          const float* __restrict__ point_w, const float* __restrict__ point_b,
                          const _Float16* __restrict__ w1h, const float* __restrict__ b1,
                          const float* __restrict__ w2, const float* __restrict__ b2,
                          const float* __restrict__ x2, const float* __restrict__ gate,
                          float* __restrict__ out) {
  __shared__ int   s_cid[4][16 * 8];
  __shared__ float s_cwt[4][16 * 8];
  __shared__ float s_fpt[4][16 * CIN];
  __shared__ __align__(64) _Float16 s_fused[4][16 * CC];
  __shared__ float s_h[4][16 * MLPH];

  const int lane = threadIdx.x & 31, wave = threadIdx.x >> 5;
  const int half = lane >> 4, ln = lane & 15;
  const int ptile = blockIdx.x * 4 + wave;        // B*N/16 tiles
  const int b = ptile / (NN / 16);
  const int p0 = (ptile % (NN / 16)) * 16;

  const float* x2b = x2 + (size_t)b * CC * SSZ;
  const float* gb  = gate + b * CC;

  if (lane < 16) {
    int n = p0 + lane;
    const float* cb = coords + (size_t)b * 3 * NN;
    float px = fminf(fmaxf(cb[n] * (float)RR - 0.5f, 0.f), (float)(RR - 1));
    float py = fminf(fmaxf(cb[NN + n] * (float)RR - 0.5f, 0.f), (float)(RR - 1));
    float pz = fminf(fmaxf(cb[2 * NN + n] * (float)RR - 0.5f, 0.f), (float)(RR - 1));
    int x0i = (int)floorf(px), y0i = (int)floorf(py), z0i = (int)floorf(pz);
    float wx = px - (float)x0i, wy = py - (float)y0i, wz = pz - (float)z0i;
    int x1i = x0i + 1 > RR - 1 ? RR - 1 : x0i + 1;
    int y1i = y0i + 1 > RR - 1 ? RR - 1 : y0i + 1;
    int z1i = z0i + 1 > RR - 1 ? RR - 1 : z0i + 1;
#pragma unroll
    for (int k = 0; k < 8; ++k) {
      int xi = (k & 4) ? x1i : x0i;
      int yi = (k & 2) ? y1i : y0i;
      int zi = (k & 1) ? z1i : z0i;
      float fw = ((k & 4) ? wx : 1.f - wx) * ((k & 2) ? wy : 1.f - wy) *
                 ((k & 1) ? wz : 1.f - wz);
      s_cid[wave][lane * 8 + k] = (xi * RR + yi) * RR + zi;
      s_cwt[wave][lane * 8 + k] = fw;
    }
#pragma unroll
    for (int i = 0; i < CIN; ++i)
      s_fpt[wave][lane * CIN + i] = feats[((size_t)b * CIN + i) * NN + n];
  }
  lds_fence();

  // fused = relu(gate*devox(x2) + relu(point MLP)) ; stage f16 [p][c]
  for (int q = lane; q < 16 * CC; q += 32) {      // uniform 32 iters/lane
    int c = q >> 4, p = q & 15;
    float ptv = 0.f;
    const float* xc = x2b + (size_t)c * SSZ;
#pragma unroll
    for (int k = 0; k < 8; ++k)
      ptv += s_cwt[wave][p * 8 + k] * xc[s_cid[wave][p * 8 + k]];
    ptv *= gb[c];
    float pt = point_b[c];
#pragma unroll
    for (int i = 0; i < CIN; ++i) pt += point_w[c * CIN + i] * s_fpt[wave][p * CIN + i];
    float f = ptv + fmaxf(pt, 0.f);
    s_fused[wave][p * CC + c] = (_Float16)fmaxf(f, 0.f);
  }
  lds_fence();

  // h = relu(W1(128x64)*fused + b1) via WMMA
  const _Float16* frow = &s_fused[wave][ln * CC];
  v16h fb0 = *(const v16h*)(frow + half * 16);
  v16h fb1 = *(const v16h*)(frow + 32 + half * 16);
#pragma unroll
  for (int mt = 0; mt < 8; ++mt) {
    const _Float16* wr = w1h + (size_t)(mt * 16 + ln) * CC;
    v16h a0 = load_a_f16(wr, half);
    v16h a1 = load_a_f16(wr + 32, half);
    v8f acc = vzero8();
    acc = wmma16(a0, fb0, acc);
    acc = wmma16(a1, fb1, acc);
#pragma unroll
    for (int r = 0; r < 8; ++r) {
      int m = mt * 16 + r + half * 8;
      s_h[wave][ln * MLPH + m] = fmaxf(acc[r] + b1[m], 0.f);
    }
  }
  lds_fence();

  // out = W2(3x128)*h + b2
  for (int q = lane; q < 16 * NCLS; q += 32) {
    int p = q / NCLS, cls = q % NCLS;
    float a = b2[cls];
    const float* wr = w2 + cls * MLPH;
    for (int m = 0; m < MLPH; ++m) a += wr[m] * s_h[wave][p * MLPH + m];
    out[((size_t)b * NCLS + cls) * NN + p0 + p] = a;
  }
}

// ---------------- launcher ----------------

extern "C" void kernel_launch(void* const* d_in, const int* in_sizes, int n_in,
                              void* d_out, int out_size, void* d_ws, size_t ws_size,
                              hipStream_t stream) {
  (void)in_sizes; (void)n_in; (void)out_size; (void)ws_size;
  const float* coords  = (const float*)d_in[0];
  const float* feats   = (const float*)d_in[1];
  const float* conv_w  = (const float*)d_in[2];
  const float* conv_b  = (const float*)d_in[3];
  const float* wq      = (const float*)d_in[4];
  const float* wk      = (const float*)d_in[5];
  const float* wv      = (const float*)d_in[6];
  const float* wo      = (const float*)d_in[7];
  const float* se_w1   = (const float*)d_in[8];
  const float* se_b1   = (const float*)d_in[9];
  const float* se_w2   = (const float*)d_in[10];
  const float* se_b2   = (const float*)d_in[11];
  const float* point_w = (const float*)d_in[12];
  const float* point_b = (const float*)d_in[13];
  const float* cls_w1  = (const float*)d_in[14];
  const float* cls_b1  = (const float*)d_in[15];
  const float* cls_w2  = (const float*)d_in[16];
  const float* cls_b2  = (const float*)d_in[17];
  float* out = (float*)d_out;

  char* ws = (char*)d_ws;
  size_t off = 0;
  auto carve = [&](size_t bytes) {
    char* p = ws + off;
    off = (off + bytes + 255) & ~(size_t)255;
    return p;
  };
  float*    g_sum = (float*)carve((size_t)BB * CIN * SSZ * 4);
  float*    g_cnt = (float*)carve((size_t)BB * SSZ * 4);
  float*    x_cs  = (float*)carve((size_t)BB * CC * SSZ * 4);
  _Float16* xT    = (_Float16*)carve((size_t)BB * SSZ * CC * 2);
  _Float16* qT    = (_Float16*)carve((size_t)BB * SSZ * CC * 2);
  _Float16* kT    = (_Float16*)carve((size_t)BB * SSZ * CC * 2);
  _Float16* v_cs  = (_Float16*)carve((size_t)BB * CC * SSZ * 2);
  _Float16* pvb   = (_Float16*)carve((size_t)BB * SSZ * CC * 2);
  float*    x2    = (float*)carve((size_t)BB * CC * SSZ * 4);
  float*    se_s  = (float*)carve((size_t)BB * CC * 4);
  float*    gateb = (float*)carve((size_t)BB * CC * 4);
  _Float16* wqh   = (_Float16*)carve((size_t)CC * CC * 2);
  _Float16* wkh   = (_Float16*)carve((size_t)CC * CC * 2);
  _Float16* wvh   = (_Float16*)carve((size_t)CC * CC * 2);
  _Float16* woh   = (_Float16*)carve((size_t)CC * CC * 2);
  _Float16* w1h   = (_Float16*)carve((size_t)MLPH * CC * 2);

  pvk_zero<<<(BB * CIN * SSZ + 255) / 256, 256, 0, stream>>>(g_sum, BB * CIN * SSZ);
  pvk_zero<<<(BB * SSZ + 255) / 256, 256, 0, stream>>>(g_cnt, BB * SSZ);
  pvk_zero<<<1, 256, 0, stream>>>(se_s, BB * CC);

  pvk_f32tof16<<<(CC * CC + 255) / 256, 256, 0, stream>>>(wq, wqh, CC * CC);
  pvk_f32tof16<<<(CC * CC + 255) / 256, 256, 0, stream>>>(wk, wkh, CC * CC);
  pvk_f32tof16<<<(CC * CC + 255) / 256, 256, 0, stream>>>(wv, wvh, CC * CC);
  pvk_f32tof16<<<(CC * CC + 255) / 256, 256, 0, stream>>>(wo, woh, CC * CC);
  pvk_f32tof16<<<(MLPH * CC + 255) / 256, 256, 0, stream>>>(cls_w1, w1h, MLPH * CC);

  pvk_scatter<<<BB * NN / 256, 256, 0, stream>>>(coords, feats, g_sum, g_cnt);
  pvk_mean<<<BB * CIN * SSZ / 256, 256, 0, stream>>>(g_sum, g_cnt);
  pvk_conv<<<BB * CC * SSZ / 256, 256, 0, stream>>>(g_sum, conv_w, conv_b, x_cs, xT);
  pvk_qkv<<<BB * SSZ / 16 / 8, 256, 0, stream>>>(wqh, wkh, wvh, xT, qT, kT, v_cs);
  pvk_flash<<<BB * SSZ / 16 / 8, 256, 0, stream>>>(qT, kT, v_cs, pvb);
  pvk_attnout<<<BB * SSZ / 16 / 8, 256, 0, stream>>>(woh, pvb, x_cs, x2, se_s);
  pvk_gate<<<1, 256, 0, stream>>>(se_s, se_w1, se_b1, se_w2, se_b2, gateb);
  pvk_point<<<BB * NN / 16 / 4, 128, 0, stream>>>(coords, feats, point_w, point_b,
                                                  w1h, cls_b1, cls_w2, cls_b2,
                                                  x2, gateb, out);
}
